// TemporalCrossAttention_48799418417193
// MI455X (gfx1250) — compile-verified, matrix operand order checked by reading, not run on hardware
//
#include <hip/hip_runtime.h>
#include <hip/hip_bf16.h>

typedef __attribute__((ext_vector_type(16))) __bf16 v16bf;
typedef __attribute__((ext_vector_type(8)))  __bf16 v8bf;
typedef __attribute__((ext_vector_type(8)))  float  v8f;

#define CDIM 256

__device__ __forceinline__ __bf16 f2bf(float f) { return (__bf16)f; }  // hw cvt, RTNE

__device__ __forceinline__ v16bf cat16(v8bf lo, v8bf hi) {
  return __builtin_shufflevector(lo, hi, 0, 1, 2, 3, 4, 5, 6, 7,
                                 8, 9, 10, 11, 12, 13, 14, 15);
}

__device__ __forceinline__ v8f wmma_bf16(v16bf a, v16bf b, v8f c) {
  // D = A(16x32 bf16) * B(32x16 bf16) + C(16x16 f32)
  return __builtin_amdgcn_wmma_f32_16x16x32_bf16(false, a, false, b, (short)0, c,
                                                 false, false);
}

// A/B fragment from a row-major bf16 row: K pattern k=(j&7)+(j>>3)*16+kh*8
__device__ __forceinline__ v16bf frag_bf16(const __bf16* row, int k0, int kh) {
  v8bf lo = *(const v8bf*)(row + k0 + kh * 8);
  v8bf hi = *(const v8bf*)(row + k0 + 16 + kh * 8);
  return cat16(lo, hi);
}

// ---------------------------------------------------------------------------
// One-shot f32 -> bf16 conversion (weights and activations).
// ---------------------------------------------------------------------------
__global__ void cvt_kernel(const float* __restrict__ src, __bf16* __restrict__ dst) {
  size_t i = (size_t)(blockIdx.x * blockDim.x + threadIdx.x) * 8;
  v8f v = *(const v8f*)(src + i);
  *(v8bf*)(dst + i) = __builtin_convertvector(v, v8bf);
}

// ---------------------------------------------------------------------------
// out = (Xb @ Wb^T + bias) * oscale ; Xb [rows,256] bf16, Wb [256,256] bf16.
// trans==0: out[row*256+col] ; trans==1: out[col*ldt+row] (for V^T)
// 1 wave per 16x16 tile.
// ---------------------------------------------------------------------------
__global__ void proj_kernel(const __bf16* __restrict__ Xb, const __bf16* __restrict__ Wb,
                            const float* __restrict__ bias, __bf16* __restrict__ out,
                            float oscale, int trans, int ldt) {
  const int lane = threadIdx.x;
  const int ln = lane & 15, kh = lane >> 4;
  const __bf16* arow = Xb + (size_t)(blockIdx.x * 16 + ln) * CDIM;
  const __bf16* brow = Wb + (size_t)(blockIdx.y * 16 + ln) * CDIM;
  v8f acc = {};
#pragma unroll
  for (int k0 = 0; k0 < CDIM; k0 += 32)
    acc = wmma_bf16(frag_bf16(arow, k0, kh), frag_bf16(brow, k0, kh), acc);

  const int col = blockIdx.y * 16 + ln;
  const float b = bias[col];
#pragma unroll
  for (int r = 0; r < 8; ++r) {
    int row = blockIdx.x * 16 + r + 8 * kh;
    float v = (acc[r] + b) * oscale;
    if (!trans) out[(size_t)row * CDIM + col] = f2bf(v);
    else        out[(size_t)col * ldt + row] = f2bf(v);
  }
}

// ---------------------------------------------------------------------------
// Flash attention: 1 wave handles 16 query rows over all M (tiles of 32).
// Q is pre-scaled by 1/sqrt(C). Emits feat(bf16) to ws, flow/conf to d_out.
// ---------------------------------------------------------------------------
__global__ void flash_kernel(const __bf16* __restrict__ Q, const __bf16* __restrict__ K,
                             const __bf16* __restrict__ Vt,
                             const float* __restrict__ kpred,
                             const float* __restrict__ kact,
                             const int* __restrict__ iterp,
                             __bf16* __restrict__ featb,
                             float* __restrict__ flow, float* __restrict__ conf,
                             int M) {
  __shared__ __bf16 ldsP[16 * 32];           // wave-private P transpose staging
  const int lane = threadIdx.x;
  const int ln = lane & 15, kh = lane >> 4;
  const int qb = blockIdx.x * 16;

  const int it = *iterp;
  const float win = (it == 0) ? 64.0f : fmaxf(4.0f, ldexpf(32.0f, 1 - it));
  const float av_ref = kact[1];              // kpts_t1_actual[0,1]

  const __bf16* qrow = Q + (size_t)(qb + ln) * CDIM;

  float pu[8], pv[8], dvok[8];
#pragma unroll
  for (int r = 0; r < 8; ++r) {
    int row = qb + r + 8 * kh;
    pu[r] = kpred[2 * row];
    pv[r] = kpred[2 * row + 1];
    dvok[r] = (fabsf(pv[r] - av_ref) < win) ? 1.0f : 0.0f;
  }

  float mi[8], li[8], pkU[8], pkV[8], anym[8];
#pragma unroll
  for (int r = 0; r < 8; ++r) { mi[r] = -3.0e38f; li[r] = 0.f; pkU[r] = 0.f; pkV[r] = 0.f; anym[r] = 0.f; }
  v8f acc[16];
#pragma unroll
  for (int c = 0; c < 16; ++c) acc[c] = (v8f){};

  for (int m0 = 0; m0 < M; m0 += 32) {
    const __bf16* krow0 = K + (size_t)(m0 + ln) * CDIM;
    const __bf16* krow1 = K + (size_t)(m0 + 16 + ln) * CDIM;
    if (m0 + 32 < M) {                       // hint next K tile toward the caches
      __builtin_prefetch(K + (size_t)(m0 + 32 + ln) * CDIM, 0, 1);
      __builtin_prefetch(K + (size_t)(m0 + 48 + ln) * CDIM, 0, 1);
    }

    v8f s0 = {}, s1 = {};
#pragma unroll
    for (int f = 0; f < 8; ++f) {
      v16bf qa = frag_bf16(qrow, f * 32, kh);
      s0 = wmma_bf16(qa, frag_bf16(krow0, f * 32, kh), s0);
      s1 = wmma_bf16(qa, frag_bf16(krow1, f * 32, kh), s1);
    }

    // columns owned by this lane
    const float au0 = kact[2 * (m0 + ln)],      avm0 = kact[2 * (m0 + ln) + 1];
    const float au1 = kact[2 * (m0 + 16 + ln)], avm1 = kact[2 * (m0 + 16 + ln) + 1];

    float rmax[8];
#pragma unroll
    for (int r = 0; r < 8; ++r) {
      bool ok0 = (fabsf(pu[r] - au0) < win) && (dvok[r] > 0.f);
      bool ok1 = (fabsf(pu[r] - au1) < win) && (dvok[r] > 0.f);
      float x0 = ok0 ? s0[r] : -1.0e9f;      // Q pre-scaled, no mul needed
      float x1 = ok1 ? s1[r] : -1.0e9f;
      anym[r] = fmaxf(anym[r], (ok0 || ok1) ? 1.0f : 0.0f);
      s0[r] = x0; s1[r] = x1;
      rmax[r] = fmaxf(x0, x1);
    }
    // row-max across the 16 lanes owning a row (xor within 16-lane halves)
#pragma unroll
    for (int r = 0; r < 8; ++r) {
      float v = rmax[r];
      v = fmaxf(v, __shfl_xor(v, 1, 32));
      v = fmaxf(v, __shfl_xor(v, 2, 32));
      v = fmaxf(v, __shfl_xor(v, 4, 32));
      v = fmaxf(v, __shfl_xor(v, 8, 32));
      rmax[r] = v;
    }

    v8f alphav;
#pragma unroll
    for (int r = 0; r < 8; ++r) {
      float mn = fmaxf(mi[r], rmax[r]);
      float alpha = __expf(mi[r] - mn);
      mi[r] = mn;
      alphav[r] = alpha;
      float p0 = __expf(s0[r] - mn);
      float p1 = __expf(s1[r] - mn);
      float rs = p0 + p1;
      rs += __shfl_xor(rs, 1, 32);
      rs += __shfl_xor(rs, 2, 32);
      rs += __shfl_xor(rs, 4, 32);
      rs += __shfl_xor(rs, 8, 32);
      li[r]  = li[r]  * alpha + rs;
      pkU[r] = pkU[r] * alpha + p0 * au0 + p1 * au1;      // per-lane partial
      pkV[r] = pkV[r] * alpha + p0 * avm0 + p1 * avm1;
      // stage P into LDS (row-major 16x32)
      int row = r + 8 * kh;
      ldsP[row * 32 + ln]      = f2bf(p0);
      ldsP[row * 32 + 16 + ln] = f2bf(p1);
    }
    // vector rescale of the 16x256 accumulator (pairs into v_pk_mul_f32)
#pragma unroll
    for (int c = 0; c < 16; ++c) acc[c] *= alphav;

    asm volatile("s_wait_dscnt 0" ::: "memory");

    // read P back as an A fragment (16x32)
    v8bf plo = *(const v8bf*)(ldsP + ln * 32 + kh * 8);
    v8bf phi = *(const v8bf*)(ldsP + ln * 32 + 16 + kh * 8);
    v16bf pf = cat16(plo, phi);
    asm volatile("s_wait_dscnt 0" ::: "memory");

    // acc += P @ V  (V^T rows give contiguous B fragments)
#pragma unroll
    for (int c = 0; c < 16; ++c) {
      const __bf16* vrow = Vt + (size_t)(c * 16 + ln) * M + m0;
      acc[c] = wmma_bf16(pf, frag_bf16(vrow, 0, kh), acc[c]);
    }
  }

  // finish cross-lane reductions for kpts / confidence
#pragma unroll
  for (int r = 0; r < 8; ++r) {
    float u = pkU[r], v = pkV[r], a = anym[r];
    u += __shfl_xor(u, 1, 32); u += __shfl_xor(u, 2, 32);
    u += __shfl_xor(u, 4, 32); u += __shfl_xor(u, 8, 32);
    v += __shfl_xor(v, 1, 32); v += __shfl_xor(v, 2, 32);
    v += __shfl_xor(v, 4, 32); v += __shfl_xor(v, 8, 32);
    a = fmaxf(a, __shfl_xor(a, 1, 32)); a = fmaxf(a, __shfl_xor(a, 2, 32));
    a = fmaxf(a, __shfl_xor(a, 4, 32)); a = fmaxf(a, __shfl_xor(a, 8, 32));
    pkU[r] = u; pkV[r] = v; anym[r] = a;
  }

  v8f invv;
#pragma unroll
  for (int r = 0; r < 8; ++r) invv[r] = 1.0f / li[r];

#pragma unroll
  for (int c = 0; c < 16; ++c) {
    v8f t = acc[c] * invv;
#pragma unroll
    for (int r = 0; r < 8; ++r) {
      int row = qb + r + 8 * kh;
      featb[(size_t)row * CDIM + c * 16 + ln] = f2bf(t[r]);
    }
  }

  if (ln == 0) {
#pragma unroll
    for (int r = 0; r < 8; ++r) {
      int row = qb + r + 8 * kh;
      flow[2 * row]     = pkU[r] * invv[r] - pu[r];
      flow[2 * row + 1] = pkV[r] * invv[r] - pv[r];
      conf[row] = anym[r] > 0.f ? 1.0f : 0.0f;
    }
  }
}

// ---------------------------------------------------------------------------
// merged = feat(bf16) @ Wmb^T + bm -> f32
// ---------------------------------------------------------------------------
__global__ void merge_kernel(const __bf16* __restrict__ A, const __bf16* __restrict__ Wb,
                             const float* __restrict__ bias, float* __restrict__ out) {
  const int lane = threadIdx.x;
  const int ln = lane & 15, kh = lane >> 4;
  const __bf16* arow = A  + (size_t)(blockIdx.x * 16 + ln) * CDIM;
  const __bf16* brow = Wb + (size_t)(blockIdx.y * 16 + ln) * CDIM;
  v8f acc = {};
#pragma unroll
  for (int k0 = 0; k0 < CDIM; k0 += 32)
    acc = wmma_bf16(frag_bf16(arow, k0, kh), frag_bf16(brow, k0, kh), acc);

  const int col = blockIdx.y * 16 + ln;
  const float b = bias[col];
#pragma unroll
  for (int r = 0; r < 8; ++r) {
    int row = blockIdx.x * 16 + r + 8 * kh;
    out[(size_t)row * CDIM + col] = acc[r] + b;
  }
}

extern "C" void kernel_launch(void* const* d_in, const int* in_sizes, int n_in,
                              void* d_out, int out_size, void* d_ws, size_t ws_size,
                              hipStream_t stream) {
  const float* nodes_t   = (const float*)d_in[0];
  const float* nodes_t1  = (const float*)d_in[1];
  const float* kpred     = (const float*)d_in[2];
  const float* kact      = (const float*)d_in[3];
  const int*   iterp     = (const int*)d_in[4];
  const float* Wq = (const float*)d_in[5];  const float* bq = (const float*)d_in[6];
  const float* Wk = (const float*)d_in[7];  const float* bk = (const float*)d_in[8];
  const float* Wv = (const float*)d_in[9];  const float* bv = (const float*)d_in[10];
  const float* Wm = (const float*)d_in[11]; const float* bm = (const float*)d_in[12];

  const int N = in_sizes[0] / CDIM;
  const int M = in_sizes[1] / CDIM;
  const size_t WSZ = (size_t)CDIM * CDIM;

  __bf16* Qb  = (__bf16*)d_ws;
  __bf16* Kb  = Qb + (size_t)N * CDIM;
  __bf16* Vt  = Kb + (size_t)M * CDIM;          // stored transposed [C, M]
  __bf16* Fb  = Vt + (size_t)M * CDIM;
  __bf16* Wqb = Fb + (size_t)N * CDIM;          // pre-converted bf16 weights
  __bf16* Wkb = Wqb + WSZ;
  __bf16* Wvb = Wkb + WSZ;
  __bf16* Wmb = Wvb + WSZ;
  __bf16* Xtb  = Wmb + WSZ;                     // pre-converted bf16 activations
  __bf16* Xt1b = Xtb + (size_t)N * CDIM;

  float* out  = (float*)d_out;
  float* flow = out + (size_t)N * CDIM;
  float* conf = flow + (size_t)N * 2;

  dim3 wave(32, 1, 1);
  const int cvtW = (int)(WSZ / (256 * 8));
  cvt_kernel<<<cvtW, 256, 0, stream>>>(Wq, Wqb);
  cvt_kernel<<<cvtW, 256, 0, stream>>>(Wk, Wkb);
  cvt_kernel<<<cvtW, 256, 0, stream>>>(Wv, Wvb);
  cvt_kernel<<<cvtW, 256, 0, stream>>>(Wm, Wmb);
  cvt_kernel<<<(int)((size_t)N * CDIM / (256 * 8)), 256, 0, stream>>>(nodes_t,  Xtb);
  cvt_kernel<<<(int)((size_t)M * CDIM / (256 * 8)), 256, 0, stream>>>(nodes_t1, Xt1b);

  const float qscale = 0.0625f;                 // 1/sqrt(256), folded into Q
  proj_kernel<<<dim3(N / 16, CDIM / 16), wave, 0, stream>>>(Xtb,  Wqb, bq, Qb, qscale, 0, 0);
  proj_kernel<<<dim3(M / 16, CDIM / 16), wave, 0, stream>>>(Xt1b, Wkb, bk, Kb, 1.0f,   0, 0);
  proj_kernel<<<dim3(M / 16, CDIM / 16), wave, 0, stream>>>(Xt1b, Wvb, bv, Vt, 1.0f,   1, M);
  flash_kernel<<<dim3(N / 16), wave, 0, stream>>>(Qb, Kb, Vt, kpred, kact, iterp,
                                                  Fb, flow, conf, M);
  merge_kernel<<<dim3(N / 16, CDIM / 16), wave, 0, stream>>>(Fb, Wmb, bm, out);
}
